// LUT3DFieldHashEncoding_58188216926693
// MI455X (gfx1250) — matrix operand order repaired and checked by reading, not compiled, not run on hardware
//
#include <hip/hip_runtime.h>
#include <hip/hip_bf16.h>
#include <math.h>

typedef __bf16 bf16;
typedef __attribute__((ext_vector_type(16))) __bf16 v16bf;
typedef __attribute__((ext_vector_type(8)))  float  v8f;

#define TABLE_SIZE (1u << 19)
#define HMASK (TABLE_SIZE - 1u)
#define HP1 2654435761u
#define HP2 805459861u
#define SAMPLES 48
#define ROWS 64   // padded rows per block

// floor(16 * (2048/16)^(l/15)) precomputed (float64 math, matches np.floor)
__constant__ float c_scales[16] = {
    16.f, 22.f, 30.f, 42.f, 58.f, 80.f, 111.f, 153.f,
    212.f, 294.f, 406.f, 561.f, 776.f, 1072.f, 1482.f, 2048.f};

// Weight region offsets (bf16 elements) inside workspace / LDS
#define W1T_OFF 0      // 64 x 32  (n-major, k contiguous)
#define W2T_OFF 2048   // 16 x 64
#define H1T_OFF 3072   // 64 x 32
#define H2T_OFF 5120   // 64 x 64
#define H3T_OFF 9216   // 16 x 64 (cols 3..15 zero)
#define W_TOTAL 10240
// bias offsets (float elements)
#define B1_OFF 0
#define B2_OFF 64
#define HB1_OFF 80
#define HB2_OFF 144
#define HB3_OFF 208
#define BIAS_TOTAL 224

struct SharedMem {
    alignas(16) bf16 W[W_TOTAL];
    float bias[BIAS_TOTAL];
    alignas(16) bf16 bufA[ROWS * 64];
    alignas(16) bf16 bufB[ROWS * 64];
    float rgb[ROWS * 3];
    float den[SAMPLES];
    float Rinv[9];
    float tinv[3];
    float sh[16];
};

// ---- weight prep: f32 row-major -> bf16 transposed (n-major) in workspace ----
__global__ void prep_weights(const float* __restrict__ w1, const float* __restrict__ w2,
                             const float* __restrict__ h1, const float* __restrict__ h2,
                             const float* __restrict__ h3,
                             const float* __restrict__ b1, const float* __restrict__ b2,
                             const float* __restrict__ hb1, const float* __restrict__ hb2,
                             const float* __restrict__ hb3,
                             bf16* __restrict__ gW, float* __restrict__ gBias) {
    int i = blockIdx.x * blockDim.x + threadIdx.x;
    if (i < W_TOTAL) {
        float v;
        if (i < W2T_OFF)       { int j = i;            int n = j / 32, k = j % 32; v = w1[k * 64 + n]; }
        else if (i < H1T_OFF)  { int j = i - W2T_OFF;  int n = j / 64, k = j % 64; v = w2[k * 16 + n]; }
        else if (i < H2T_OFF)  { int j = i - H1T_OFF;  int n = j / 32, k = j % 32; v = h1[k * 64 + n]; }
        else if (i < H3T_OFF)  { int j = i - H2T_OFF;  int n = j / 64, k = j % 64; v = h2[k * 64 + n]; }
        else                   { int j = i - H3T_OFF;  int n = j / 64, k = j % 64; v = (n < 3) ? h3[k * 3 + n] : 0.0f; }
        gW[i] = (bf16)v;
    }
    if (i < BIAS_TOTAL) {
        float v;
        if (i < B2_OFF)        v = b1[i];
        else if (i < HB1_OFF)  v = b2[i - B2_OFF];
        else if (i < HB2_OFF)  v = hb1[i - HB1_OFF];
        else if (i < HB3_OFF)  v = hb2[i - HB2_OFF];
        else                   v = (i - HB3_OFF < 3) ? hb3[i - HB3_OFF] : 0.0f;
        gBias[i] = v;
    }
}

// ---- fragment loaders (wave32 WMMA operands; contiguous k order, same in A and B) ----
// A (16x32 bf16): lanes 0-15 hold M=lane, K=0..15; lanes 16-31 hold M=lane-16, K=16..31
__device__ __forceinline__ v16bf load_a_frag(const bf16* lds, int strip, int stride, int ktile, int lane) {
    int m = strip * 16 + (lane & 15);
    int hl = lane >> 4;
    const bf16* p = lds + m * stride + ktile * 32 + hl * 16;
    union { v16bf v; uint4 q[2]; } f;
    f.q[0] = *(const uint4*)(p);
    f.q[1] = *(const uint4*)(p + 8);
    return f.v;
}
// B (32x16 bf16, K x N): lanes 0-15 hold N=lane, K=0..15; lanes 16-31 hold N=lane-16, K=16..31
// Wt stored n-major (row n = output column, K contiguous) -> 32B contiguous per lane
__device__ __forceinline__ v16bf load_b_frag(const bf16* Wt, int K, int ctile, int ktile, int lane) {
    int n = ctile * 16 + (lane & 15);
    int kb = ktile * 32 + (lane >> 4) * 16;
    const bf16* p = Wt + n * K + kb;
    union { v16bf v; uint4 q[2]; } f;
    f.q[0] = *(const uint4*)(p);
    f.q[1] = *(const uint4*)(p + 8);
    return f.v;
}

// One dense layer on this wave's 16-row strip: out = act(in[16xK] @ W[KxN] + bias)
// A fragments are loaded once and reused across all column tiles.
template <int K, int NCT, bool SIGMOID>
__device__ __forceinline__ void mlp_layer(const bf16* ldsIn, int strideIn,
                                          const bf16* Wt, const float* bias,
                                          bf16* ldsOut, int strideOut,
                                          float* rgbOut, int strip, int lane) {
    constexpr int KT = K / 32;
    v16bf a[KT];
#pragma unroll
    for (int kt = 0; kt < KT; ++kt) a[kt] = load_a_frag(ldsIn, strip, strideIn, kt, lane);
#pragma unroll
    for (int ct = 0; ct < NCT; ++ct) {
        v8f acc = {0.f, 0.f, 0.f, 0.f, 0.f, 0.f, 0.f, 0.f};
#pragma unroll
        for (int kt = 0; kt < KT; ++kt) {
            v16bf w = load_b_frag(Wt, K, ct, kt, lane);
            acc = __builtin_amdgcn_wmma_f32_16x16x32_bf16(false, a[kt], false, w, (short)0, acc,
                                                          false, false);
        }
        // C/D layout: VGPR i -> row strip*16 + i + 8*(lane>=16), col = ct*16 + (lane&15)
        int col = ct * 16 + (lane & 15);
        float bv = bias[col];
        int rbase = strip * 16 + (lane >> 4) * 8;
#pragma unroll
        for (int i = 0; i < 8; ++i) {
            float v = acc[i] + bv;
            if (SIGMOID) {
                v = 1.0f / (1.0f + __expf(-v));
                if ((lane & 15) < 3) rgbOut[(rbase + i) * 3 + (lane & 15)] = v;
            } else {
                v = fmaxf(v, 0.0f);
                ldsOut[(rbase + i) * strideOut + col] = (bf16)v;
            }
        }
    }
}

__global__ __launch_bounds__(128) void nerf_fused(
    const float* __restrict__ positions, const float* __restrict__ densities,
    const float* __restrict__ normals, const float* __restrict__ c2w,
    const float* __restrict__ table, const bf16* __restrict__ gW,
    const float* __restrict__ gBias, float* __restrict__ out) {
    __shared__ SharedMem sm;
    const int tid = threadIdx.x;
    const int lane = tid & 31;
    const int strip = tid >> 5;
    const int b = blockIdx.x;

    // ---- stage weights + biases + densities ----
    {
        const uint4* src = (const uint4*)gW;
        uint4* dst = (uint4*)sm.W;
        for (int i = tid; i < W_TOTAL / 8; i += 128) dst[i] = src[i];
        for (int i = tid; i < BIAS_TOTAL; i += 128) sm.bias[i] = gBias[i];
    }
    if (tid < SAMPLES) sm.den[tid] = densities[b * SAMPLES + tid];
    if (tid == 0) {
        // Rinv = R^T; tinv = -Rinv @ t; n_cam = Rinv @ n -> (n+1)/2 -> SH16
        float Rm[3][4];
#pragma unroll
        for (int i = 0; i < 3; ++i)
#pragma unroll
            for (int j = 0; j < 4; ++j) Rm[i][j] = c2w[b * 12 + i * 4 + j];
#pragma unroll
        for (int i = 0; i < 3; ++i) {
            sm.Rinv[i * 3 + 0] = Rm[0][i];
            sm.Rinv[i * 3 + 1] = Rm[1][i];
            sm.Rinv[i * 3 + 2] = Rm[2][i];
            sm.tinv[i] = -(Rm[0][i] * Rm[0][3] + Rm[1][i] * Rm[1][3] + Rm[2][i] * Rm[2][3]);
        }
        float n0 = normals[b * 3 + 0], n1 = normals[b * 3 + 1], n2 = normals[b * 3 + 2];
        float x = (Rm[0][0] * n0 + Rm[1][0] * n1 + Rm[2][0] * n2 + 1.0f) * 0.5f;
        float y = (Rm[0][1] * n0 + Rm[1][1] * n1 + Rm[2][1] * n2 + 1.0f) * 0.5f;
        float z = (Rm[0][2] * n0 + Rm[1][2] * n1 + Rm[2][2] * n2 + 1.0f) * 0.5f;
        float xx = x * x, yy = y * y, zz = z * z;
        sm.sh[0] = 0.28209479177387814f;
        sm.sh[1] = -0.48860251190291987f * y;
        sm.sh[2] = 0.48860251190291987f * z;
        sm.sh[3] = -0.48860251190291987f * x;
        sm.sh[4] = 1.0925484305920792f * x * y;
        sm.sh[5] = -1.0925484305920792f * y * z;
        sm.sh[6] = 0.94617469575756f * zz - 0.31539156525252f;
        sm.sh[7] = -1.0925484305920792f * x * z;
        sm.sh[8] = 0.5462742152960396f * (xx - yy);
        sm.sh[9] = 0.5900435899266435f * y * (3.0f * xx - yy);
        sm.sh[10] = 2.890611442640554f * x * y * z;
        sm.sh[11] = 0.4570457994644657f * y * (5.0f * zz - 1.0f);
        sm.sh[12] = 0.37317633259011546f * z * (5.0f * zz - 3.0f);
        sm.sh[13] = 0.4570457994644657f * x * (5.0f * zz - 1.0f);
        sm.sh[14] = 1.445305721320277f * z * (xx - yy);
        sm.sh[15] = 0.5900435899266435f * x * (xx - 3.0f * yy);
    }
    __syncthreads();

    // ---- hash encoding: thread t -> sample r = t/2, 8 levels; row r is inside this wave's strip ----
    {
        const int r = tid >> 1;
        const int lb = (tid & 1) * 8;
        if (r < SAMPLES) {
            const float* p = positions + (size_t)(b * SAMPLES + r) * 3;
            float px = p[0], py = p[1], pz = p[2];
            float x = sm.Rinv[0] * px + sm.Rinv[1] * py + sm.Rinv[2] * pz + sm.tinv[0];
            float y = sm.Rinv[3] * px + sm.Rinv[4] * py + sm.Rinv[5] * pz + sm.tinv[1];
            float z = sm.Rinv[6] * px + sm.Rinv[7] * py + sm.Rinv[8] * pz + sm.tinv[2];
            x = (x + 1.0f) * 0.5f; y = (y + 1.0f) * 0.5f; z = (z + 1.0f) * 0.5f;
            bool sel = (x > 0.f) && (x < 1.f) && (y > 0.f) && (y < 1.f) && (z > 0.f) && (z < 1.f);
            float s = sel ? 1.0f : 0.0f;
            x *= s; y *= s; z *= s;
            const float2* T = (const float2*)table;
#pragma unroll
            for (int li = 0; li < 8; ++li) {
                int l = lb + li;
                float sc = c_scales[l];
                float sx = x * sc, sy = y * sc, sz = z * sc;
                float fx = floorf(sx), fy = floorf(sy), fz = floorf(sz);
                float cx = ceilf(sx), cy = ceilf(sy), cz = ceilf(sz);
                float ox = sx - fx, oy = sy - fy, oz = sz - fz;
                unsigned fxi = (unsigned)fx, fyi = (unsigned)fy, fzi = (unsigned)fz;
                unsigned cxi = (unsigned)cx, cyi = (unsigned)cy, czi = (unsigned)cz;
                unsigned base = (unsigned)l * TABLE_SIZE;
                auto H = [&](unsigned X, unsigned Y, unsigned Z) {
                    return ((X ^ (Y * HP1) ^ (Z * HP2)) & HMASK) + base;
                };
                float2 f0 = T[H(cxi, cyi, czi)], f1 = T[H(cxi, fyi, czi)];
                float2 f2 = T[H(fxi, fyi, czi)], f3 = T[H(fxi, cyi, czi)];
                float2 f4 = T[H(cxi, cyi, fzi)], f5 = T[H(cxi, fyi, fzi)];
                float2 f6 = T[H(fxi, fyi, fzi)], f7 = T[H(fxi, cyi, fzi)];
                float e0, e1;
                {
                    float a03 = f0.x * ox + f3.x * (1.f - ox);
                    float a12 = f1.x * ox + f2.x * (1.f - ox);
                    float a56 = f5.x * ox + f6.x * (1.f - ox);
                    float a47 = f4.x * ox + f7.x * (1.f - ox);
                    e0 = (a03 * oy + a12 * (1.f - oy)) * oz + (a47 * oy + a56 * (1.f - oy)) * (1.f - oz);
                    float b03 = f0.y * ox + f3.y * (1.f - ox);
                    float b12 = f1.y * ox + f2.y * (1.f - ox);
                    float b56 = f5.y * ox + f6.y * (1.f - ox);
                    float b47 = f4.y * ox + f7.y * (1.f - ox);
                    e1 = (b03 * oy + b12 * (1.f - oy)) * oz + (b47 * oy + b56 * (1.f - oy)) * (1.f - oz);
                }
                union { bf16 h[2]; unsigned u; } pk;
                pk.h[0] = (bf16)e0; pk.h[1] = (bf16)e1;
                *(unsigned*)&sm.bufA[r * 32 + l * 2] = pk.u;
            }
        } else {
#pragma unroll
            for (int li = 0; li < 8; ++li)
                *(unsigned*)&sm.bufA[r * 32 + (lb + li) * 2] = 0u;
        }
    }
    // Each wave only reads rows it wrote (per-wave LDS ordering) -> no barrier needed.

    // ---- geo MLP ----
    mlp_layer<32, 4, false>(sm.bufA, 32, sm.W + W1T_OFF, sm.bias + B1_OFF, sm.bufB, 64, nullptr, strip, lane); // enc->64
    // fill SH dir features into bufA cols 0..15 (enc in bufA is dead now)
    if (lane < 16) {
        bf16 dv = (bf16)sm.sh[lane];
#pragma unroll
        for (int r = 0; r < 16; ++r) sm.bufA[(strip * 16 + r) * 32 + lane] = dv;
    }
    mlp_layer<64, 1, false>(sm.bufB, 64, sm.W + W2T_OFF, sm.bias + B2_OFF, sm.bufA + 16, 32, nullptr, strip, lane); // geo -> bufA cols 16..31

    // ---- color MLP ----
    mlp_layer<32, 4, false>(sm.bufA, 32, sm.W + H1T_OFF, sm.bias + HB1_OFF, sm.bufB, 64, nullptr, strip, lane);
    mlp_layer<64, 4, false>(sm.bufB, 64, sm.W + H2T_OFF, sm.bias + HB2_OFF, sm.bufA, 64, nullptr, strip, lane);
    mlp_layer<64, 1, true>(sm.bufA, 64, sm.W + H3T_OFF, sm.bias + HB3_OFF, nullptr, 0, sm.rgb, strip, lane);

    __syncthreads();

    // ---- softmax over S, weighted RGB sum (3 threads, one per channel) ----
    if (tid < 3) {
        int c = tid;
        float m = -3.4e38f;
        for (int s2 = 0; s2 < SAMPLES; ++s2) m = fmaxf(m, sm.den[s2]);
        float sum = 0.f, acc = 0.f;
        for (int s2 = 0; s2 < SAMPLES; ++s2) {
            float e = __expf(sm.den[s2] - m);
            sum += e;
            acc += e * sm.rgb[s2 * 3 + c];
        }
        out[b * 3 + c] = acc / sum;
    }
}

extern "C" void kernel_launch(void* const* d_in, const int* in_sizes, int n_in,
                              void* d_out, int out_size, void* d_ws, size_t ws_size,
                              hipStream_t stream) {
    (void)n_in; (void)out_size; (void)ws_size;
    const float* positions = (const float*)d_in[0];
    const float* densities = (const float*)d_in[1];
    const float* normals   = (const float*)d_in[2];
    const float* c2w       = (const float*)d_in[3];
    const float* table     = (const float*)d_in[4];
    const float* w1  = (const float*)d_in[5];
    const float* b1  = (const float*)d_in[6];
    const float* w2  = (const float*)d_in[7];
    const float* b2  = (const float*)d_in[8];
    const float* h1  = (const float*)d_in[9];
    const float* hb1 = (const float*)d_in[10];
    const float* h2  = (const float*)d_in[11];
    const float* hb2 = (const float*)d_in[12];
    const float* h3  = (const float*)d_in[13];
    const float* hb3 = (const float*)d_in[14];

    const int B = in_sizes[2] / 3;  // normals: (B,3)

    bf16*  gW    = (bf16*)d_ws;                              // 10240 bf16 = 20480 B
    float* gBias = (float*)((char*)d_ws + W_TOTAL * 2);      // 224 floats

    prep_weights<<<(W_TOTAL + 255) / 256, 256, 0, stream>>>(w1, w2, h1, h2, h3,
                                                            b1, b2, hb1, hb2, hb3, gW, gBias);
    nerf_fused<<<B, 128, 0, stream>>>(positions, densities, normals, c2w, table, gW, gBias,
                                      (float*)d_out);
}
